// LiZAttention_48215302864936
// MI455X (gfx1250) — matrix operand order repaired and checked by reading, not compiled, not run on hardware
//
#include <hip/hip_runtime.h>
#include <hip/hip_bf16.h>
#include <stdint.h>

typedef __attribute__((ext_vector_type(16))) __bf16 v16bf;
typedef __attribute__((ext_vector_type(8)))  float  v8f;

#define HEADS  16
#define DH     128
#define DMODEL 2048
#define NCTX   1024
#define BATCH  2
#define CHUNK  64

// Native fp32 -> bf16 convert (RNE on gfx1250; hardware v_cvt, not a bit-twiddle)
__device__ __forceinline__ __bf16 f2bf(float f) { return (__bf16)f; }

__device__ __forceinline__ int kmapA(int e, int h) {
  return (e < 8) ? (8 * h + e) : (16 + 8 * h + (e - 8));
}

__device__ __forceinline__ v8f zero8() {
  v8f z;
#pragma unroll
  for (int i = 0; i < 8; ++i) z[i] = 0.0f;
  return z;
}

__device__ __forceinline__ v8f wmma_bf16(v16bf a, v16bf b, v8f c) {
  return __builtin_amdgcn_wmma_f32_16x16x32_bf16(false, a, false, b, (short)0, c,
                                                 false, false);
}

// A fragment (16xK tile): a[e] = A[row][kb + kmapA(e,h)], from bf16 LDS
__device__ __forceinline__ v16bf fragA_lds(const __bf16* p, int ld, int row, int kb, int h) {
  v16bf a;
#pragma unroll
  for (int e = 0; e < 16; ++e) a[e] = p[row * ld + kb + kmapA(e, h)];
  return a;
}
// A fragment from f32 LDS (convert)
__device__ __forceinline__ v16bf fragA_ldsf(const float* p, int ld, int row, int kb, int h) {
  v16bf a;
#pragma unroll
  for (int e = 0; e < 16; ++e) a[e] = f2bf(p[row * ld + kb + kmapA(e, h)]);
  return a;
}
// A fragment from global f32
__device__ __forceinline__ v16bf fragA_g(const float* p, int ld, int row, int kb, int h) {
  v16bf a;
#pragma unroll
  for (int e = 0; e < 16; ++e) a[e] = f2bf(p[(size_t)row * ld + kb + kmapA(e, h)]);
  return a;
}
// B fragment: lane holds N = col + lr, elements e = K = kb + kmapA(e,h). stored row-major [k][n]
__device__ __forceinline__ v16bf fragB_lds(const __bf16* p, int ld, int kb, int col, int h, int lr) {
  v16bf b;
#pragma unroll
  for (int e = 0; e < 16; ++e) b[e] = p[(kb + kmapA(e, h)) * ld + col + lr];
  return b;
}
// B fragment where B = stored^T: B[k][n] = stored[n][k]
__device__ __forceinline__ v16bf fragBT_lds(const __bf16* p, int ld, int kb, int col, int h, int lr) {
  v16bf b;
#pragma unroll
  for (int e = 0; e < 16; ++e) b[e] = p[(col + lr) * ld + kb + kmapA(e, h)];
  return b;
}

// ---------------------------------------------------------------------------
// GEMM: C[M,N] = A[M,K] @ B[K,N], fp32 in/out, bf16 WMMA inside.
// Block 256 thr = 8 waves (2x4); 128x128 tile; BK=32.
// ---------------------------------------------------------------------------
__global__ __launch_bounds__(256) void gemm_bf16(const float* __restrict__ A,
                                                 const float* __restrict__ B,
                                                 float* __restrict__ C,
                                                 int M, int N, int K) {
  __shared__ __bf16 sA[128 * 34];
  __shared__ __bf16 sB[32 * 130];
  const int tid = threadIdx.x;
  const int lane = tid & 31, wid = tid >> 5;
  const int h = lane >> 4, lr = lane & 15;
  const int wm = wid >> 2, wn = wid & 3;
  const int m0 = blockIdx.y * 128, n0 = blockIdx.x * 128;

  v8f acc[4][2];
#pragma unroll
  for (int mt = 0; mt < 4; ++mt)
#pragma unroll
    for (int nt = 0; nt < 2; ++nt) acc[mt][nt] = zero8();

  for (int k0 = 0; k0 < K; k0 += 32) {
#pragma unroll
    for (int it = 0; it < 4; ++it) {
      int idx = tid + it * 256;
      {
        int r = idx >> 3, c = (idx & 7) * 4;
        float4 v = *(const float4*)(A + (size_t)(m0 + r) * K + k0 + c);
        sA[r * 34 + c + 0] = f2bf(v.x); sA[r * 34 + c + 1] = f2bf(v.y);
        sA[r * 34 + c + 2] = f2bf(v.z); sA[r * 34 + c + 3] = f2bf(v.w);
      }
      {
        int r = idx >> 5, c = (idx & 31) * 4;
        float4 v = *(const float4*)(B + (size_t)(k0 + r) * N + n0 + c);
        sB[r * 130 + c + 0] = f2bf(v.x); sB[r * 130 + c + 1] = f2bf(v.y);
        sB[r * 130 + c + 2] = f2bf(v.z); sB[r * 130 + c + 3] = f2bf(v.w);
      }
    }
    if (k0 + 32 < K) {
      __builtin_prefetch(A + (size_t)(m0 + (tid >> 3)) * K + k0 + 32, 0, 1);
      __builtin_prefetch(B + (size_t)(k0 + 32 + (tid >> 5)) * N + n0 + (tid & 31) * 4, 0, 1);
    }
    __syncthreads();

    v16bf af[4], bfr[2];
#pragma unroll
    for (int mt = 0; mt < 4; ++mt)
      af[mt] = fragA_lds(sA, 34, wm * 64 + mt * 16 + lr, 0, h);
#pragma unroll
    for (int nt = 0; nt < 2; ++nt)
      bfr[nt] = fragB_lds(sB, 130, 0, wn * 32 + nt * 16, h, lr);
#pragma unroll
    for (int mt = 0; mt < 4; ++mt)
#pragma unroll
      for (int nt = 0; nt < 2; ++nt)
        acc[mt][nt] = wmma_bf16(af[mt], bfr[nt], acc[mt][nt]);
    __syncthreads();
  }
#pragma unroll
  for (int mt = 0; mt < 4; ++mt)
#pragma unroll
    for (int nt = 0; nt < 2; ++nt)
#pragma unroll
      for (int r = 0; r < 8; ++r)
        C[(size_t)(m0 + wm * 64 + mt * 16 + r + 8 * h) * N + n0 + wn * 32 + nt * 16 + lr] =
            acc[mt][nt][r];
}

// ---------------------------------------------------------------------------
// Base causal softmax attention (flash style). Grid (n/64, H, B); 128 threads.
// Each wave owns 16 query rows; online softmax via half-wave shuffles.
// ---------------------------------------------------------------------------
__global__ __launch_bounds__(128) void base_attn(const float* __restrict__ q,
                                                 const float* __restrict__ k,
                                                 const float* __restrict__ v,
                                                 float* __restrict__ o) {
  __shared__ __bf16 sQ[64 * 130];
  __shared__ __bf16 sKb[64 * 130];
  __shared__ __bf16 sV[64 * 130];
  __shared__ __bf16 sP[4 * 16 * 72];
  const int tid = threadIdx.x, lane = tid & 31, wid = tid >> 5;
  const int h = lane >> 4, lr = lane & 15;
  const int qb = blockIdx.x, head = blockIdx.y, b = blockIdx.z;
  const size_t base = (size_t)b * NCTX * DMODEL + (size_t)head * DH;
  const int ld = DMODEL;
  const float scale = 0.0883883476483184f; // 1/sqrt(128)

#pragma unroll
  for (int it = 0; it < 16; ++it) {
    int idx = tid + it * 128;
    int r = idx >> 5, c = (idx & 31) * 4;
    float4 vv = *(const float4*)(q + base + (size_t)(qb * 64 + r) * ld + c);
    sQ[r * 130 + c + 0] = f2bf(vv.x * scale); sQ[r * 130 + c + 1] = f2bf(vv.y * scale);
    sQ[r * 130 + c + 2] = f2bf(vv.z * scale); sQ[r * 130 + c + 3] = f2bf(vv.w * scale);
  }

  float m_r[8], l_r[8];
  v8f O[8];
#pragma unroll
  for (int r = 0; r < 8; ++r) { m_r[r] = -1e30f; l_r[r] = 0.0f; }
#pragma unroll
  for (int dt = 0; dt < 8; ++dt) O[dt] = zero8();
  __bf16* myP = sP + wid * 16 * 72;

  for (int kb0 = 0; kb0 <= qb; ++kb0) {
    __syncthreads();
#pragma unroll
    for (int it = 0; it < 16; ++it) {
      int idx = tid + it * 128;
      int r = idx >> 5, c = (idx & 31) * 4;
      float4 kv = *(const float4*)(k + base + (size_t)(kb0 * 64 + r) * ld + c);
      float4 vv = *(const float4*)(v + base + (size_t)(kb0 * 64 + r) * ld + c);
      sKb[r * 130 + c + 0] = f2bf(kv.x); sKb[r * 130 + c + 1] = f2bf(kv.y);
      sKb[r * 130 + c + 2] = f2bf(kv.z); sKb[r * 130 + c + 3] = f2bf(kv.w);
      sV[r * 130 + c + 0] = f2bf(vv.x); sV[r * 130 + c + 1] = f2bf(vv.y);
      sV[r * 130 + c + 2] = f2bf(vv.z); sV[r * 130 + c + 3] = f2bf(vv.w);
    }
    __syncthreads();

    v8f Sc[4];
#pragma unroll
    for (int nt = 0; nt < 4; ++nt) Sc[nt] = zero8();
#pragma unroll
    for (int ks = 0; ks < 4; ++ks) {
      v16bf a = fragA_lds(sQ, 130, wid * 16 + lr, ks * 32, h);
#pragma unroll
      for (int nt = 0; nt < 4; ++nt) {
        v16bf bb = fragBT_lds(sKb, 130, ks * 32, nt * 16, h, lr);
        Sc[nt] = wmma_bf16(a, bb, Sc[nt]);
      }
    }
#pragma unroll
    for (int r = 0; r < 8; ++r) {
      int rowg = qb * 64 + wid * 16 + r + 8 * h;
      float mx = m_r[r];
#pragma unroll
      for (int nt = 0; nt < 4; ++nt) {
        int colg = kb0 * 64 + nt * 16 + lr;
        float s = (colg <= rowg) ? Sc[nt][r] : -1e30f;
        Sc[nt][r] = s;
        mx = fmaxf(mx, s);
      }
#pragma unroll
      for (int off = 1; off < 16; off <<= 1) mx = fmaxf(mx, __shfl_xor(mx, off, 32));
      float alpha = __expf(m_r[r] - mx);
      float psum = 0.0f;
#pragma unroll
      for (int nt = 0; nt < 4; ++nt) {
        float p = __expf(Sc[nt][r] - mx);
        myP[(r + 8 * h) * 72 + nt * 16 + lr] = f2bf(p);
        psum += p;
      }
#pragma unroll
      for (int off = 1; off < 16; off <<= 1) psum += __shfl_xor(psum, off, 32);
      l_r[r] = l_r[r] * alpha + psum;
      m_r[r] = mx;
#pragma unroll
      for (int dt = 0; dt < 8; ++dt) O[dt][r] *= alpha;
    }
#pragma unroll
    for (int ks = 0; ks < 2; ++ks) {
      v16bf a = fragA_lds(myP, 72, lr, ks * 32, h);
#pragma unroll
      for (int dt = 0; dt < 8; ++dt) {
        v16bf bb = fragB_lds(sV, 130, ks * 32, dt * 16, h, lr);
        O[dt] = wmma_bf16(a, bb, O[dt]);
      }
    }
  }
#pragma unroll
  for (int dt = 0; dt < 8; ++dt)
#pragma unroll
    for (int r = 0; r < 8; ++r) {
      int rowg = qb * 64 + wid * 16 + r + 8 * h;
      o[base + (size_t)rowg * ld + dt * 16 + lr] = O[dt][r] / l_r[r];
    }
}

// ---------------------------------------------------------------------------
// Prep: per-head softmax+clip for q,k; gate g; q *= dk^-0.5; v*=g; kb=k*g.
// One wave per (b,l,h) segment of 128; each lane holds 4 values.
// ---------------------------------------------------------------------------
__global__ __launch_bounds__(256) void prep_kernel(float* __restrict__ Q, float* __restrict__ K,
                                                   float* __restrict__ V, float* __restrict__ KB,
                                                   int total) {
  int seg = blockIdx.x * 8 + (threadIdx.x >> 5);
  if (seg >= total) return;
  int lane = threadIdx.x & 31;
  size_t off = (size_t)seg * DH + lane * 4;
  float4 qv = *(const float4*)(Q + off);
  float4 kv = *(const float4*)(K + off);
  float4 vv = *(const float4*)(V + off);
  const float lo = 1e-6f, hi = 1.0f - 1e-6f, qsc = 0.0883883476483184f;

  // gate from raw k: clip(logsigmoid(k)/16, -16, 16)
  auto lsig = [](float x) { return fminf(x, 0.0f) - log1pf(__expf(-fabsf(x))); };
  auto gate = [&](float x) { return fminf(fmaxf(lsig(x) * 0.0625f, -16.0f), 16.0f); };
  float4 g; g.x = gate(kv.x); g.y = gate(kv.y); g.z = gate(kv.z); g.w = gate(kv.w);

  // softmax(q) clipped, scaled
  float m = fmaxf(fmaxf(qv.x, qv.y), fmaxf(qv.z, qv.w));
#pragma unroll
  for (int o2 = 1; o2 < 32; o2 <<= 1) m = fmaxf(m, __shfl_xor(m, o2, 32));
  float4 e; e.x = __expf(qv.x - m); e.y = __expf(qv.y - m); e.z = __expf(qv.z - m); e.w = __expf(qv.w - m);
  float s = e.x + e.y + e.z + e.w;
#pragma unroll
  for (int o2 = 1; o2 < 32; o2 <<= 1) s += __shfl_xor(s, o2, 32);
  float inv = 1.0f / s;
  qv.x = fminf(fmaxf(e.x * inv, lo), hi) * qsc; qv.y = fminf(fmaxf(e.y * inv, lo), hi) * qsc;
  qv.z = fminf(fmaxf(e.z * inv, lo), hi) * qsc; qv.w = fminf(fmaxf(e.w * inv, lo), hi) * qsc;

  // softmax(k) clipped
  m = fmaxf(fmaxf(kv.x, kv.y), fmaxf(kv.z, kv.w));
#pragma unroll
  for (int o2 = 1; o2 < 32; o2 <<= 1) m = fmaxf(m, __shfl_xor(m, o2, 32));
  e.x = __expf(kv.x - m); e.y = __expf(kv.y - m); e.z = __expf(kv.z - m); e.w = __expf(kv.w - m);
  s = e.x + e.y + e.z + e.w;
#pragma unroll
  for (int o2 = 1; o2 < 32; o2 <<= 1) s += __shfl_xor(s, o2, 32);
  inv = 1.0f / s;
  kv.x = fminf(fmaxf(e.x * inv, lo), hi); kv.y = fminf(fmaxf(e.y * inv, lo), hi);
  kv.z = fminf(fmaxf(e.z * inv, lo), hi); kv.w = fminf(fmaxf(e.w * inv, lo), hi);

  float4 vb; vb.x = vv.x * g.x; vb.y = vv.y * g.y; vb.z = vv.z * g.z; vb.w = vv.w * g.w;
  float4 kbv; kbv.x = kv.x * g.x; kbv.y = kv.y * g.y; kbv.z = kv.z * g.z; kbv.w = kv.w * g.w;
  *(float4*)(Q + off) = qv;
  *(float4*)(K + off) = kv;
  *(float4*)(V + off) = vb;
  *(float4*)(KB + off) = kbv;
}

// ---------------------------------------------------------------------------
// Chunked delta rule. One WG (8 waves) per (b,h). State S (128x128 fp32) lives
// in WMMA accumulators (wave w owns rows 16w..16w+15); staged to LDS bf16 in
// halves only when consumed as a B operand. LDS = exactly 64KB.
// ---------------------------------------------------------------------------
__global__ __launch_bounds__(256) void delta_kernel(const float* __restrict__ q,
                                                    const float* __restrict__ k,
                                                    const float* __restrict__ v,
                                                    const float* __restrict__ kb,
                                                    float* __restrict__ o) {
  __shared__ __bf16 sK[64 * 128];  // k chunk
  __shared__ __bf16 sU[64 * 128];  // kb -> (-w) -> u
  __shared__ float  sT[64 * 64];   // L -> T -> attn
  __shared__ __bf16 sS[64 * 128];  // vb staging / S half staging
  const int tid = threadIdx.x, lane = tid & 31, wid = tid >> 5;
  const int h = lane >> 4, lr = lane & 15;
  const int bh = blockIdx.x;
  const int b = bh / HEADS, head = bh % HEADS;
  const size_t base = (size_t)b * NCTX * DMODEL + (size_t)head * DH;
  const int ld = DMODEL;
  const int umt = wid & 3;
  const int unt0 = (wid >> 2) * 4;
  float* sRow = (float*)sS;  // 64-float overlay, used only in phase 2

  v8f S[8];
#pragma unroll
  for (int nt = 0; nt < 8; ++nt) S[nt] = zero8();

  for (int c0 = 0; c0 < NCTX; c0 += CHUNK) {
    // phase 0: stage k, kb chunks as bf16
#pragma unroll
    for (int it = 0; it < 8; ++it) {
      int idx = tid + it * 256;
      int r = idx >> 5, c = (idx & 31) * 4;
      float4 kv = *(const float4*)(k + base + (size_t)(c0 + r) * ld + c);
      float4 bv = *(const float4*)(kb + base + (size_t)(c0 + r) * ld + c);
      sK[r * 128 + c + 0] = f2bf(kv.x); sK[r * 128 + c + 1] = f2bf(kv.y);
      sK[r * 128 + c + 2] = f2bf(kv.z); sK[r * 128 + c + 3] = f2bf(kv.w);
      sU[r * 128 + c + 0] = f2bf(bv.x); sU[r * 128 + c + 1] = f2bf(bv.y);
      sU[r * 128 + c + 2] = f2bf(bv.z); sU[r * 128 + c + 3] = f2bf(bv.w);
    }
    __syncthreads();

    // phase 1: L = strict_tril(kb @ k^T) -> sT fp32
#pragma unroll
    for (int j = 0; j < 2; ++j) {
      int t = wid * 2 + j, mt = t >> 2, nt = t & 3;
      v8f L = zero8();
#pragma unroll
      for (int ks = 0; ks < 4; ++ks) {
        v16bf a = fragA_lds(sU, 128, mt * 16 + lr, ks * 32, h);
        v16bf bb = fragBT_lds(sK, 128, ks * 32, nt * 16, h, lr);
        L = wmma_bf16(a, bb, L);
      }
#pragma unroll
      for (int r = 0; r < 8; ++r) {
        int row = mt * 16 + r + 8 * h, col = nt * 16 + lr;
        sT[row * 64 + col] = (col < row) ? L[r] : 0.0f;
      }
    }
    __syncthreads();

    // phase 2: in-place T = (I+L)^{-1}, unit lower triangular
    for (int i = 1; i < 64; ++i) {
      if (tid < i) sRow[tid] = sT[i * 64 + tid];
      __syncthreads();
      if (tid < i) {
        float acc = sRow[tid];
        for (int p = tid + 1; p < i; ++p) acc = fmaf(sRow[p], sT[p * 64 + tid], acc);
        sT[i * 64 + tid] = -acc;
      }
      __syncthreads();
    }
    if (tid < 64) sT[tid * 64 + tid] = 1.0f;
    __syncthreads();

    // phase 3: w = T @ kb ; store -w into sU
    v8f W[4];
#pragma unroll
    for (int j = 0; j < 4; ++j) W[j] = zero8();
#pragma unroll
    for (int ks = 0; ks < 2; ++ks) {
      v16bf a = fragA_ldsf(sT, 64, umt * 16 + lr, ks * 32, h);
#pragma unroll
      for (int j = 0; j < 4; ++j) {
        v16bf bb = fragB_lds(sU, 128, ks * 32, (unt0 + j) * 16, h, lr);
        W[j] = wmma_bf16(a, bb, W[j]);
      }
    }
    __syncthreads();
#pragma unroll
    for (int j = 0; j < 4; ++j)
#pragma unroll
      for (int r = 0; r < 8; ++r)
        sU[(umt * 16 + r + 8 * h) * 128 + (unt0 + j) * 16 + lr] = f2bf(-W[j][r]);
    __syncthreads();

    // phase 4: u = T@vb + (-w)@S
#pragma unroll
    for (int it = 0; it < 8; ++it) {  // stage vb into sS
      int idx = tid + it * 256;
      int r = idx >> 5, c = (idx & 31) * 4;
      float4 vv = *(const float4*)(v + base + (size_t)(c0 + r) * ld + c);
      sS[r * 128 + c + 0] = f2bf(vv.x); sS[r * 128 + c + 1] = f2bf(vv.y);
      sS[r * 128 + c + 2] = f2bf(vv.z); sS[r * 128 + c + 3] = f2bf(vv.w);
    }
    __syncthreads();
    v8f U[4];
#pragma unroll
    for (int j = 0; j < 4; ++j) U[j] = zero8();
#pragma unroll
    for (int ks = 0; ks < 2; ++ks) {
      v16bf a = fragA_ldsf(sT, 64, umt * 16 + lr, ks * 32, h);
#pragma unroll
      for (int j = 0; j < 4; ++j) {
        v16bf bb = fragB_lds(sS, 128, ks * 32, (unt0 + j) * 16, h, lr);
        U[j] = wmma_bf16(a, bb, U[j]);
      }
    }
    __syncthreads();
#pragma unroll
    for (int half = 0; half < 2; ++half) {  // U += (-w) @ S, staged halves
      if ((wid >> 2) == half) {
#pragma unroll
        for (int nt = 0; nt < 8; ++nt)
#pragma unroll
          for (int r = 0; r < 8; ++r)
            sS[(16 * (wid & 3) + r + 8 * h) * 128 + nt * 16 + lr] = f2bf(S[nt][r]);
      }
      __syncthreads();
#pragma unroll
      for (int ks = 0; ks < 2; ++ks) {
        v16bf a = fragA_lds(sU, 128, umt * 16 + lr, half * 64 + ks * 32, h);
#pragma unroll
        for (int j = 0; j < 4; ++j) {
          v16bf bb = fragB_lds(sS, 128, ks * 32, (unt0 + j) * 16, h, lr);
          U[j] = wmma_bf16(a, bb, U[j]);
        }
      }
      __syncthreads();
    }
#pragma unroll
    for (int j = 0; j < 4; ++j)  // store u into sU (replacing -w)
#pragma unroll
      for (int r = 0; r < 8; ++r)
        sU[(umt * 16 + r + 8 * h) * 128 + (unt0 + j) * 16 + lr] = f2bf(U[j][r]);
    __syncthreads();

    // phase 5: attn = incl_tril(q @ k^T) -> sT fp32 (T is dead)
#pragma unroll
    for (int j = 0; j < 2; ++j) {
      int t = wid * 2 + j, mt = t >> 2, nt = t & 3;
      v8f Aa = zero8();
#pragma unroll
      for (int ks = 0; ks < 4; ++ks) {
        v16bf a = fragA_g(q + base, ld, c0 + mt * 16 + lr, ks * 32, h);
        v16bf bb = fragBT_lds(sK, 128, ks * 32, nt * 16, h, lr);
        Aa = wmma_bf16(a, bb, Aa);
      }
#pragma unroll
      for (int r = 0; r < 8; ++r) {
        int row = mt * 16 + r + 8 * h, col = nt * 16 + lr;
        sT[row * 64 + col] = (col <= row) ? Aa[r] : 0.0f;
      }
    }
    __syncthreads();

    // phase 6: o = attn@u + q@S
    v8f O[4];
#pragma unroll
    for (int j = 0; j < 4; ++j) O[j] = zero8();
#pragma unroll
    for (int ks = 0; ks < 2; ++ks) {
      v16bf a = fragA_ldsf(sT, 64, umt * 16 + lr, ks * 32, h);
#pragma unroll
      for (int j = 0; j < 4; ++j) {
        v16bf bb = fragB_lds(sU, 128, ks * 32, (unt0 + j) * 16, h, lr);
        O[j] = wmma_bf16(a, bb, O[j]);
      }
    }
#pragma unroll
    for (int half = 0; half < 2; ++half) {
      if ((wid >> 2) == half) {
#pragma unroll
        for (int nt = 0; nt < 8; ++nt)
#pragma unroll
          for (int r = 0; r < 8; ++r)
            sS[(16 * (wid & 3) + r + 8 * h) * 128 + nt * 16 + lr] = f2bf(S[nt][r]);
      }
      __syncthreads();
#pragma unroll
      for (int ks = 0; ks < 2; ++ks) {
        v16bf a = fragA_g(q + base, ld, c0 + umt * 16 + lr, half * 64 + ks * 32, h);
#pragma unroll
        for (int j = 0; j < 4; ++j) {
          v16bf bb = fragB_lds(sS, 128, ks * 32, (unt0 + j) * 16, h, lr);
          O[j] = wmma_bf16(a, bb, O[j]);
        }
      }
      __syncthreads();
    }
#pragma unroll
    for (int j = 0; j < 4; ++j)
#pragma unroll
      for (int r = 0; r < 8; ++r)
        o[base + (size_t)(c0 + umt * 16 + r + 8 * h) * ld + (unt0 + j) * 16 + lr] = O[j][r];

    // phase 7: S += k^T @ u (accumulate directly in register state)
#pragma unroll
    for (int ks = 0; ks < 2; ++ks) {
      v16bf a;  // A[m][kk] = k[kk][16*wid + m]
#pragma unroll
      for (int e = 0; e < 16; ++e) a[e] = sK[(ks * 32 + kmapA(e, h)) * 128 + 16 * wid + lr];
#pragma unroll
      for (int nt = 0; nt < 8; ++nt) {
        v16bf bb = fragB_lds(sU, 128, ks * 32, nt * 16, h, lr);
        S[nt] = wmma_bf16(a, bb, S[nt]);
      }
    }
    __syncthreads();
  }
}

__global__ __launch_bounds__(256) void mix_kernel(float* __restrict__ olin,
                                                  const float* __restrict__ obase, int n4) {
  int i = blockIdx.x * 256 + threadIdx.x;
  if (i < n4) {
    float4 a = *(const float4*)(olin + (size_t)i * 4);
    float4 bb = *(const float4*)(obase + (size_t)i * 4);
    a.x = 0.5f * a.x + 0.5f * bb.x; a.y = 0.5f * a.y + 0.5f * bb.y;
    a.z = 0.5f * a.z + 0.5f * bb.z; a.w = 0.5f * a.w + 0.5f * bb.w;
    *(float4*)(olin + (size_t)i * 4) = a;
  }
}

extern "C" void kernel_launch(void* const* d_in, const int* in_sizes, int n_in,
                              void* d_out, int out_size, void* d_ws, size_t ws_size,
                              hipStream_t stream) {
  (void)in_sizes; (void)n_in; (void)out_size; (void)ws_size;
  const float* x  = (const float*)d_in[0];
  const float* Wq = (const float*)d_in[1];
  const float* Wk = (const float*)d_in[2];
  const float* Wv = (const float*)d_in[3];
  const float* Wo = (const float*)d_in[4];
  float* out = (float*)d_out;

  const int M = BATCH * NCTX;               // 2048
  const size_t tsz = (size_t)M * DMODEL;    // 4M elements
  float* qb = (float*)d_ws;
  float* kbuf = qb + tsz;
  float* vb = kbuf + tsz;
  float* kbb = vb + tsz;
  float* olin = kbb + tsz;
  float* obase = olin + tsz;

  dim3 gg(DMODEL / 128, M / 128);
  gemm_bf16<<<gg, 256, 0, stream>>>(x, Wq, qb, M, DMODEL, DMODEL);
  gemm_bf16<<<gg, 256, 0, stream>>>(x, Wk, kbuf, M, DMODEL, DMODEL);
  gemm_bf16<<<gg, 256, 0, stream>>>(x, Wv, vb, M, DMODEL, DMODEL);

  base_attn<<<dim3(NCTX / 64, HEADS, BATCH), 128, 0, stream>>>(qb, kbuf, vb, obase);

  prep_kernel<<<(BATCH * NCTX * HEADS + 7) / 8, 256, 0, stream>>>(qb, kbuf, vb, kbb,
                                                                  BATCH * NCTX * HEADS);

  delta_kernel<<<BATCH * HEADS, 256, 0, stream>>>(qb, kbuf, vb, kbb, olin);

  mix_kernel<<<((int)(tsz / 4) + 255) / 256, 256, 0, stream>>>(olin, obase, (int)(tsz / 4));

  gemm_bf16<<<gg, 256, 0, stream>>>(olin, Wo, out, M, DMODEL, DMODEL);
}
